// PoolingAttention_3195455668462
// MI455X (gfx1250) — compile-verified, matrix operand order checked by reading, not run on hardware
//
#include <hip/hip_runtime.h>
#include <math.h>

typedef __attribute__((ext_vector_type(16))) __bf16 v16bf;
typedef __attribute__((ext_vector_type(8)))  __bf16 v8bf;
typedef __attribute__((ext_vector_type(8)))  float  v8f;
typedef __attribute__((ext_vector_type(4)))  unsigned int v4u;
typedef __attribute__((ext_vector_type(8)))  int v8i;
typedef __attribute__((ext_vector_type(4)))  int v4i;

#define S_DIM 4096
#define F_DIM 1024
#define A_DIM 512

// ---------------------------------------------------------------------------
// Small helper kernels for the algebraically-collapsed attention branch
// ---------------------------------------------------------------------------

__global__ __launch_bounds__(256) void k_mean_sum(const float* __restrict__ inp,
                                                  float* __restrict__ meanSum,
                                                  int S, int F) {
  int f = blockIdx.x * 256 + threadIdx.x;
  int s0 = blockIdx.y * 512;
  int b = blockIdx.z;
  const float* p = inp + ((size_t)b * S + s0) * F + f;
  float acc = 0.f;
  for (int j = 0; j < 512; ++j) acc += p[(size_t)j * F];
  atomicAdd(&meanSum[b * F + f], acc);
}

__global__ __launch_bounds__(256) void k_agg(const float* __restrict__ W,
                                             const float* __restrict__ bias,
                                             const float* __restrict__ meanSum,
                                             float* __restrict__ agg,
                                             int F, int S) {
  int id = blockIdx.x * 256 + threadIdx.x;   // B*512 threads
  int b = id >> 9, a = id & 511;
  const float* wr = W + (size_t)a * F;
  const float* mr = meanSum + b * F;
  float s = 0.f;
  for (int f = 0; f < F; ++f) s += wr[f] * mr[f];
  agg[id] = s * (1.0f / (float)S) + bias[a];
}

__global__ __launch_bounds__(256) void k_keyvec(const float* __restrict__ W,
                                                const float* __restrict__ bias,
                                                const float* __restrict__ agg,
                                                float* __restrict__ keyvec,
                                                float* __restrict__ keybias,
                                                int F) {
  int b = blockIdx.x;
  int tid = threadIdx.x;
  const float* ag = agg + b * 512;
  float acc[4] = {0.f, 0.f, 0.f, 0.f};
  for (int a = 0; a < 512; ++a) {
    float av = ag[a];
    const float* wr = W + (size_t)(512 + a) * F;
#pragma unroll
    for (int i = 0; i < 4; ++i) acc[i] += av * wr[tid + i * 256];
  }
#pragma unroll
  for (int i = 0; i < 4; ++i) keyvec[b * F + tid + i * 256] = acc[i];
  if (tid == 0) {
    float s = 0.f;
    for (int a = 0; a < 512; ++a) s += ag[a] * bias[512 + a];
    keybias[b] = s;
  }
}

__global__ __launch_bounds__(256) void k_scores(const float* __restrict__ inp,
                                                const float* __restrict__ keyvec,
                                                const float* __restrict__ keybias,
                                                float* __restrict__ scores,
                                                int S, int F) {
  int b = blockIdx.y;
  int wave = threadIdx.x >> 5, lane = threadIdx.x & 31;
  int s = blockIdx.x * 8 + wave;
  const float* row = inp + ((size_t)b * S + s) * F;
  const float* kv = keyvec + b * F;
  float acc = 0.f;
  for (int i = lane; i < F; i += 32) acc += row[i] * kv[i];
#pragma unroll
  for (int m = 16; m >= 1; m >>= 1) acc += __shfl_xor(acc, m, 32);
  if (lane == 0) scores[b * S + s] = acc + keybias[b];
}

__global__ __launch_bounds__(1024) void k_softmax(float* __restrict__ scores, int S) {
  int b = blockIdx.x, tid = threadIdx.x;
  __shared__ float red[1024];
  float* sc = scores + b * S;
  float v[4];
  float m = -INFINITY;
#pragma unroll
  for (int i = 0; i < 4; ++i) { v[i] = sc[tid + i * 1024]; m = fmaxf(m, v[i]); }
  red[tid] = m;
  __syncthreads();
  for (int st = 512; st >= 1; st >>= 1) {
    if (tid < st) red[tid] = fmaxf(red[tid], red[tid + st]);
    __syncthreads();
  }
  m = red[0];
  __syncthreads();
  float sum = 0.f;
#pragma unroll
  for (int i = 0; i < 4; ++i) { v[i] = __expf(v[i] - m); sum += v[i]; }
  red[tid] = sum;
  __syncthreads();
  for (int st = 512; st >= 1; st >>= 1) {
    if (tid < st) red[tid] += red[tid + st];
    __syncthreads();
  }
  float inv = 1.0f / red[0];
#pragma unroll
  for (int i = 0; i < 4; ++i) sc[tid + i * 1024] = v[i] * inv;
}

__global__ __launch_bounds__(256) void k_wmean(const float* __restrict__ inp,
                                               const float* __restrict__ w,
                                               float* __restrict__ wmean,
                                               int S, int F) {
  int f = blockIdx.x * 256 + threadIdx.x;
  int s0 = blockIdx.y * 512;
  int b = blockIdx.z;
  const float* p = inp + ((size_t)b * S + s0) * F + f;
  const float* wp = w + b * S + s0;
  float acc = 0.f;
  for (int j = 0; j < 512; ++j) acc += wp[j] * p[(size_t)j * F];
  atomicAdd(&wmean[b * F + f], acc);
}

__global__ __launch_bounds__(256) void k_ctx(const float* __restrict__ W,
                                             const float* __restrict__ bias,
                                             const float* __restrict__ wmean,
                                             float* __restrict__ ctx,
                                             int F) {
  int id = blockIdx.x * 256 + threadIdx.x;
  int b = id >> 9, a = id & 511;
  const float* wr = W + (size_t)(1024 + a) * F;
  const float* mr = wmean + b * F;
  float s = 0.f;
  for (int f = 0; f < F; ++f) s += wr[f] * mr[f];
  ctx[id] = s + bias[1024 + a];
}

// Pre-convert the seg/loc/out W rows to bf16, permuted so that each 64-column
// tile's 192 rows (3 chunks x 64) are contiguous -> TDM-friendly 2D tensor.
__global__ __launch_bounds__(256) void k_wconv(const float* __restrict__ W,
                                               __bf16* __restrict__ Wbf, int F) {
  int row = blockIdx.x;                 // 0..1535
  int ctile = row / 192, j = row % 192;
  int wrow = 1536 + (j >> 6) * 512 + ctile * 64 + (j & 63);
  const float* src = W + (size_t)wrow * F;
  __bf16* dst = Wbf + (size_t)row * F;
  for (int k = threadIdx.x; k < F; k += 256) dst[k] = (__bf16)src[k];
}

// ---------------------------------------------------------------------------
// Fused WMMA GEMM (seg/loc/out projections) + segment-max + sliding-window max
// + aggregated combine. One block per (batch, 64-row segment, 64-col tile).
// B operand staged into LDS by the Tensor Data Mover (TDM).
// ---------------------------------------------------------------------------

#define ASTR 72   // bf16 elements per A row in LDS (64 + 8 pad, keeps 16B align)
#define BSTR 72   // matches TDM pad: 128B row + 16B pad
#define ESTR 68   // f32 elements per epilogue row (64 + 4 pad)
#define KC   64   // K-chunk staged per iteration (two K=32 wmma steps)
#define BS_LDS_OFF 11520u   // byte offset of Bs within the (single) LDS block

__device__ __forceinline__ v16bf ldfrag(const __bf16* p) {
  // elements [0..7] = K..K+7, elements [8..15] = K+16..K+23 (per ISA A/B layout)
  v8bf lo = *(const v8bf*)(p);
  v8bf hi = *(const v8bf*)(p + 16);
  v16bf r;
#pragma unroll
  for (int i = 0; i < 8; ++i) { r[i] = lo[i]; r[i + 8] = hi[i]; }
  return r;
}

// Issue one TDM load: 192x64 bf16 tile (row stride 1024 elems) -> LDS at
// BS_LDS_OFF, hardware-padded 16B per 128B row (=> BSTR*2 byte LDS rows).
__device__ __forceinline__ void tdm_load_B(const __bf16* gsrc) {
  unsigned long long ga = (unsigned long long)(uintptr_t)gsrc;
  v4u g0;
  g0[0] = 1u;                                   // count=1, user descriptor
  g0[1] = BS_LDS_OFF;                           // lds_addr (bytes)
  g0[2] = (unsigned)(ga & 0xFFFFFFFFu);         // global_addr[31:0]
  g0[3] = (unsigned)((ga >> 32) & 0x01FFFFFFu)  // global_addr[56:32]
        | (2u << 30);                           // type=2 ("image")
  v8i g1;
  g1[0] = (int)((1u << 16)    // data_size = 2 bytes
              | (1u << 20)    // pad_enable
              | (4u << 22)    // pad_interval: 32 DWORDs = 128 B
              | (3u << 25));  // pad_amount: 4 DWORDs = 16 B
  g1[1] = (int)(1024u << 16); // [47:32] atomic_barrier=0 | [63:48] tensor_dim0.lo
  g1[2] = (int)(192u << 16);  // [79:64] tensor_dim0.hi=0 | [95:80] tensor_dim1.lo
  g1[3] = (int)(64u << 16);   // [111:96] tensor_dim1.hi=0 | [127:112] tile_dim0=64
  g1[4] = 192;                // [143:128] tile_dim1=192 | [159:144] tile_dim2=0
  g1[5] = 1024;               // [191:160] tensor_dim0_stride.lo = 1024
  g1[6] = 0;                  // stride.hi | tensor_dim1_stride.lo
  g1[7] = 0;
  v4i zz = {0, 0, 0, 0};
#if defined(__clang_major__) && (__clang_major__ >= 23)
  v8i zz8 = {0, 0, 0, 0, 0, 0, 0, 0};
  __builtin_amdgcn_tensor_load_to_lds(g0, g1, zz, zz, zz8, 0);
#else
  __builtin_amdgcn_tensor_load_to_lds(g0, g1, zz, zz, 0);
#endif
}

__global__ __launch_bounds__(256) void k_fused(const float* __restrict__ inp,
                                               const __bf16* __restrict__ Wbf,
                                               const float* __restrict__ bias,
                                               const float* __restrict__ ctx,
                                               float* __restrict__ out,
                                               int S, int F) {
  const int ctile = blockIdx.x;          // column tile of 64 within A=512
  const int segid = blockIdx.y;          // 64-row segment
  const int b = blockIdx.z;
  const int s0 = segid * 64;
  const int c0 = ctile * 64;
  const int tid = threadIdx.x;
  const int wave = tid >> 5;
  const int lane = tid & 31;
  const int laneRow = lane & 15;
  const int kb = (lane >> 4) << 3;       // lanes 0-15: K base 0; lanes 16-31: 8

  __shared__ __align__(16) unsigned char smem[57344];
  __bf16* As = (__bf16*)smem;                     // [80][ASTR]  rows s0-8..s0+71
  __bf16* Bs = (__bf16*)(smem + BS_LDS_OFF);      // [192][BSTR] TDM destination
  float* outS = (float*)smem;                     // [64][ESTR]  (aliases staging)
  float* segS = (float*)(smem + 17408);           // [64][ESTR]
  float* locS = (float*)(smem + 34816);           // [80][ESTR]
  float* smax = (float*)(smem + 56576);           // [64]

  const v8f vz = {0.f, 0.f, 0.f, 0.f, 0.f, 0.f, 0.f, 0.f};
  v8f acc[7];
#pragma unroll
  for (int i = 0; i < 7; ++i) acc[i] = vz;

  const __bf16* wtile = Wbf + (size_t)ctile * 192 * F;   // this tile's 192 rows

  for (int k0 = 0; k0 < F; k0 += KC) {
    __syncthreads();
    // --- stage B via Tensor Data Mover (wave 0 issues; TDM ignores EXEC) ---
    if (tid < 32) tdm_load_B(wtile + k0);
    // --- stage A: 80 rows x 64 K-values, f32 -> bf16 (needs conversion) ---
    for (int i = tid; i < 80 * 16; i += 256) {
      int row = i >> 4, q = i & 15;
      int rg = s0 - 8 + row;
      rg = rg < 0 ? 0 : (rg >= S ? S - 1 : rg);   // halo clamp; masked at use
      const float4 v = *(const float4*)(inp + ((size_t)b * S + rg) * F + k0 + q * 4);
      __bf16* d = As + row * ASTR + q * 4;
      d[0] = (__bf16)v.x; d[1] = (__bf16)v.y; d[2] = (__bf16)v.z; d[3] = (__bf16)v.w;
    }
    if (k0 + KC < F)
      __builtin_prefetch(inp + ((size_t)b * S + s0) * F + k0 + KC, 0, 1);
    if (tid < 32) __builtin_amdgcn_s_wait_tensorcnt(0);
    __syncthreads();

    // --- 52 C-tiles (out:16, seg:16, loc:20) spread over 8 waves ---
#pragma unroll
    for (int slot = 0; slot < 7; ++slot) {
      const int t = wave + slot * 8;     // wave-uniform
      if (t < 52) {
        int chunk, mbase, ntile;
        if (t < 16)      { chunk = 2; mbase = 8 + ((t >> 2) << 4); ntile = t & 3; }
        else if (t < 32) { int u = t - 16; chunk = 0; mbase = 8 + ((u >> 2) << 4); ntile = u & 3; }
        else             { int u = t - 32; chunk = 1; mbase = (u >> 2) << 4;       ntile = u & 3; }
        const __bf16* aRow = As + (mbase + laneRow) * ASTR + kb;
        const __bf16* bRow = Bs + (chunk * 64 + (ntile << 4) + laneRow) * BSTR + kb;
#pragma unroll
        for (int kk = 0; kk < KC; kk += 32) {
          v16bf afr = ldfrag(aRow + kk);
          v16bf bfr = ldfrag(bRow + kk);
          acc[slot] = __builtin_amdgcn_wmma_f32_16x16x32_bf16(
              false, afr, false, bfr, (short)0, acc[slot], false, false);
        }
      }
    }
  }
  __syncthreads();   // staging LDS dead; safe to alias with epilogue arrays

  // --- dump C tiles (+bias) into LDS epilogue arrays ---
#pragma unroll
  for (int slot = 0; slot < 7; ++slot) {
    const int t = wave + slot * 8;
    if (t < 52) {
      float* dst; int rbase, ntile, bbase;
      if (t < 16)      { dst = outS; rbase = (t >> 2) << 4; ntile = t & 3; bbase = 2560; }
      else if (t < 32) { int u = t - 16; dst = segS; rbase = (u >> 2) << 4; ntile = u & 3; bbase = 1536; }
      else             { int u = t - 32; dst = locS; rbase = (u >> 2) << 4; ntile = u & 3; bbase = 2048; }
      int col = (ntile << 4) + laneRow;
      float bv = bias[bbase + c0 + col];
      int rofs = (lane >> 4) << 3;       // C layout: VGPR r -> M=r (lo lanes) / r+8 (hi)
#pragma unroll
      for (int r = 0; r < 8; ++r)
        dst[(rbase + r + rofs) * ESTR + col] = acc[slot][r] + bv;
    }
  }
  __syncthreads();

  // --- per-column segment max over the 64 rows ---
  if (tid < 64) {
    float m = -INFINITY;
    for (int r = 0; r < 64; ++r) m = fmaxf(m, segS[r * ESTR + tid]);
    smax[tid] = m;
  }
  __syncthreads();

  // --- sliding-window max (kernel 5, -inf pad via global-bound mask) + combine ---
  const float* ctxb = ctx + b * A_DIM + c0;
  for (int i = tid; i < 64 * 64; i += 256) {
    int row = i >> 6, col = i & 63;
    float lm = -INFINITY;
#pragma unroll
    for (int d = -2; d <= 2; ++d) {
      int gs = s0 + row + d;
      if (gs >= 0 && gs < S) lm = fmaxf(lm, locS[(8 + row + d) * ESTR + col]);
    }
    float val = (ctxb[col] + smax[col]) * outS[row * ESTR + col] + lm;
    out[((size_t)b * S + s0 + row) * A_DIM + c0 + col] = val;
  }
}

// ---------------------------------------------------------------------------
extern "C" void kernel_launch(void* const* d_in, const int* in_sizes, int n_in,
                              void* d_out, int out_size, void* d_ws, size_t ws_size,
                              hipStream_t stream) {
  (void)n_in; (void)out_size; (void)ws_size;
  const int S = S_DIM, F = F_DIM;
  const float* inp  = (const float*)d_in[0];
  const float* W    = (const float*)d_in[1];
  const float* bias = (const float*)d_in[2];
  float* out = (float*)d_out;
  const int B = in_sizes[0] / (S * F);

  float* ws    = (float*)d_ws;
  float* MEAN  = ws;                        // B*F
  float* AGG   = MEAN + (size_t)B * F;      // B*512
  float* KEYV  = AGG + (size_t)B * 512;     // B*F
  float* KEYB  = KEYV + (size_t)B * F;      // B
  float* SCOR  = KEYB + B;                  // B*S (reused as softmax weights)
  float* WMEAN = SCOR + (size_t)B * S;      // B*F
  float* CTX   = WMEAN + (size_t)B * F;     // B*512
  __bf16* WBF  = (__bf16*)(CTX + (size_t)B * 512);  // 1536*F bf16 (3 MB)

  hipMemsetAsync(MEAN, 0, (size_t)B * F * sizeof(float), stream);
  hipMemsetAsync(WMEAN, 0, (size_t)B * F * sizeof(float), stream);

  k_wconv<<<dim3(1536), 256, 0, stream>>>(W, WBF, F);
  k_mean_sum<<<dim3(F / 256, S / 512, B), 256, 0, stream>>>(inp, MEAN, S, F);
  k_agg<<<dim3(B * 2), 256, 0, stream>>>(W, bias, MEAN, AGG, F, S);
  k_keyvec<<<dim3(B), 256, 0, stream>>>(W, bias, AGG, KEYV, KEYB, F);
  k_scores<<<dim3(S / 8, B), 256, 0, stream>>>(inp, KEYV, KEYB, SCOR, S, F);
  k_softmax<<<dim3(B), 1024, 0, stream>>>(SCOR, S);
  k_wmean<<<dim3(F / 256, S / 512, B), 256, 0, stream>>>(inp, SCOR, WMEAN, S, F);
  k_ctx<<<dim3(B * 2), 256, 0, stream>>>(W, bias, WMEAN, CTX, F);
  k_fused<<<dim3(A_DIM / 64, S / 64, B), 256, 0, stream>>>(inp, WBF, bias, CTX, out, S, F);
}
